// RNNWrapper_89928025244452
// MI455X (gfx1250) — matrix operand order repaired and checked by reading, not compiled, not run on hardware
//
#include <hip/hip_runtime.h>
#include <hip/hip_bf16.h>

// ---------------- problem constants ----------------
#define HD   128   // hidden / channels / state size
#define FD   128   // freq bins
#define BSZ  4
#define TD   512
#define TC   64    // time chunk
#define NCHUNK (TD / TC)
#define NTHREADS 256
#define NWAVE 8

typedef __attribute__((ext_vector_type(16))) __bf16 bf16x16;
typedef __attribute__((ext_vector_type(8)))  float  f32x8;
typedef __attribute__((ext_vector_type(4)))  unsigned int u32x4;
typedef __attribute__((ext_vector_type(8)))  int          i32x8;
typedef __attribute__((ext_vector_type(4)))  int          i32x4;

// ---------------- LDS layout (bytes) ----------------
#define OFF_WBRE 0
#define OFF_WBIM (OFF_WBRE + HD*HD*2)
#define OFF_WCRE (OFF_WBIM + HD*HD*2)
#define OFF_WCIM (OFF_WCRE + HD*HD*2)
#define OFF_XA   (OFF_WCIM + HD*HD*2)     // TC x 128 bf16 (A operand, row-major [t][c])
#define OFF_URE  (OFF_XA   + TC*HD*2)     // TC x 128 f32
#define OFF_UIM  (OFF_URE  + TC*HD*4)
#define OFF_HRE  (OFF_UIM  + TC*HD*4)     // TC x 128 bf16 (A operand for output GEMM)
#define OFF_HIM  (OFF_HRE  + TC*HD*2)     //   holds  -h_im  (minus folded in)
#define STGB     (TC*HD*4)                // one f32 staging buffer: [c][t], 32 KB
#define OFF_XSTG (OFF_HIM + TC*HD*2)      // double-buffered TDM destination
#define LDS_BYTES (OFF_XSTG + 2*STGB)     // 311296 B = 304 KB (<= 320 KB/WGP)

// f32 -> bf16, round-to-nearest-even
static __device__ __forceinline__ unsigned short f2bf(float f) {
    union { float f; unsigned u; } v; v.f = f;
    unsigned r = (v.u + 0x7FFFu + ((v.u >> 16) & 1u)) >> 16;
    return (unsigned short)r;
}

// A operand (16x32 bf16) from row-major [rows][128] bf16 LDS.
// ISA A layout: lane L -> row (L&15); v0..3 = K {k0+8*hi ..}, v4..7 = K {k0+16+8*hi ..}
static __device__ __forceinline__ bf16x16
ld_tileA(const unsigned short* base, int row0, int k0, int lane) {
    const unsigned short* p = base + (row0 + (lane & 15)) * HD + k0 + ((lane >> 4) << 3);
    bf16x16 r;
    uint4* o = reinterpret_cast<uint4*>(&r);
    o[0] = *reinterpret_cast<const uint4*>(p);
    o[1] = *reinterpret_cast<const uint4*>(p + 16);
    return r;
}

// B operand (32x16 bf16) from weights stored row-major [n][128] (= B^T column n).
// ISA B layout: lane L -> column (L&15); v0..3 = K {k0+8*hi ..}, v4..7 = K {k0+16+8*hi ..}
static __device__ __forceinline__ bf16x16
ld_tileB(const unsigned short* base, int n0, int k0, int lane) {
    const unsigned short* p = base + (n0 + (lane & 15)) * HD + k0 + ((lane >> 4) << 3);
    bf16x16 r;
    uint4* o = reinterpret_cast<uint4*>(&r);
    o[0] = *reinterpret_cast<const uint4*>(p);
    o[1] = *reinterpret_cast<const uint4*>(p + 16);
    return r;
}

// pack float4 (scaled by g) as 4 bf16 and store 8B to LDS
static __device__ __forceinline__ void
wstore(unsigned short* w, int idx4, float4 v, float g) {
    uint2 pk;
    pk.x = (unsigned)f2bf(v.x * g) | ((unsigned)f2bf(v.y * g) << 16);
    pk.y = (unsigned)f2bf(v.z * g) | ((unsigned)f2bf(v.w * g) << 16);
    reinterpret_cast<uint2*>(w)[idx4] = pk;
}

// Issue one TDM 2D tile load: x[b, c=0..127, f, t0..t0+TC) -> LDS f32 [c][t].
// D# per CDNA5 ISA 8.3/8.4: group0 = {count/type/lds/global}, group1 = dims/strides.
// This toolchain exposes the 6-arg builtin:
//   (uint32x4 g0, int32x8 g1, int32x4 g2, int32x4 g3, int32x8 extra, i32 cpol)
static __device__ __forceinline__ void
tdm_load_x_chunk(const float* xg, int b, int f, int t0, unsigned lds_off) {
    unsigned long long ga = (unsigned long long)(const void*)
        (xg + ((size_t)b * HD * FD + (size_t)f) * TD + t0);   // c = 0 row
    u32x4 g0;
    g0[0] = 1u;                                               // count=1, user mode
    g0[1] = lds_off;                                          // lds_addr (bytes)
    g0[2] = (unsigned)(ga & 0xFFFFFFFFull);                   // global_addr[31:0]
    g0[3] = (unsigned)((ga >> 32) & 0x01FFFFFFull)            // global_addr[56:32]
          | 0x80000000u;                                      // type=2 ("image")
    i32x8 g1;
    g1[0] = (int)(2u << 16);                                  // data_size = 4 B
    g1[1] = (int)(((unsigned)TD & 0xFFFFu) << 16);            // tensor_dim0 lo16 (=512)
    g1[2] = (int)((((unsigned)TD >> 16) & 0xFFFFu)            // tensor_dim0 hi16
          | (((unsigned)HD & 0xFFFFu) << 16));                // tensor_dim1 lo16 (=128)
    g1[3] = (int)((((unsigned)HD >> 16) & 0xFFFFu)            // tensor_dim1 hi16
          | ((unsigned)TC << 16));                            // tile_dim0 = 64 (t)
    g1[4] = (int)(unsigned)HD;                                // tile_dim1 = 128 (c), tile_dim2=0
    g1[5] = (int)(FD * TD);                                   // tensor_dim0_stride = 65536
    g1[6] = 0;                                                // stride0 hi16 | stride1 lo16
    g1[7] = 0;                                                // stride1 hi32 (2D: unused)
    i32x4 z4 = {0, 0, 0, 0};
    i32x8 z8 = {0, 0, 0, 0, 0, 0, 0, 0};
    __builtin_amdgcn_tensor_load_to_lds(g0, g1, z4, z4, z8, 0);
}

__global__ void __launch_bounds__(NTHREADS)
lru_fused_wmma_kernel(const float* __restrict__ xg,
                      const float* __restrict__ nu_log,
                      const float* __restrict__ theta_log,
                      const float* __restrict__ gamma_log,
                      const float* __restrict__ Bre,
                      const float* __restrict__ Bim,
                      const float* __restrict__ Cre,
                      const float* __restrict__ Cim,
                      const float* __restrict__ Dg,
                      float* __restrict__ outg) {
    extern __shared__ char smem[];
    unsigned short* wBre = reinterpret_cast<unsigned short*>(smem + OFF_WBRE);
    unsigned short* wBim = reinterpret_cast<unsigned short*>(smem + OFF_WBIM);
    unsigned short* wCre = reinterpret_cast<unsigned short*>(smem + OFF_WCRE);
    unsigned short* wCim = reinterpret_cast<unsigned short*>(smem + OFF_WCIM);
    unsigned short* xA   = reinterpret_cast<unsigned short*>(smem + OFF_XA);
    float*          uRe  = reinterpret_cast<float*>(smem + OFF_URE);
    float*          uIm  = reinterpret_cast<float*>(smem + OFF_UIM);
    unsigned short* hRe  = reinterpret_cast<unsigned short*>(smem + OFF_HRE);
    unsigned short* hIm  = reinterpret_cast<unsigned short*>(smem + OFF_HIM);

    const int tid  = threadIdx.x;
    const int wave = tid >> 5;
    const int lane = tid & 31;
    const int f    = blockIdx.x >> 2;   // bin
    const int b    = blockIdx.x & 3;    // batch

    // ---- kick off TDM for chunk 0 while we convert weights ----
    if (wave == 0)
        tdm_load_x_chunk(xg, b, f, 0, OFF_XSTG);

    // ---- one-time: convert weights to bf16 in LDS; fold gamma into B ----
    {
        const float4* br4 = reinterpret_cast<const float4*>(Bre + (size_t)f * HD * HD);
        const float4* bi4 = reinterpret_cast<const float4*>(Bim + (size_t)f * HD * HD);
        const float4* cr4 = reinterpret_cast<const float4*>(Cre + (size_t)f * HD * HD);
        const float4* ci4 = reinterpret_cast<const float4*>(Cim + (size_t)f * HD * HD);
        for (int idx = tid; idx < HD * HD / 4; idx += NTHREADS) {
            int n   = idx >> 5;                       // weight row
            float g = __expf(gamma_log[f * HD + n]);  // u = gamma * (B x)
            wstore(wBre, idx, br4[idx], g);
            wstore(wBim, idx, bi4[idx], g);
            wstore(wCre, idx, cr4[idx], 1.0f);
            wstore(wCim, idx, ci4[idx], 1.0f);
        }
    }

    // ---- per-state recurrence parameters (threads 0..127 own state n=tid) ----
    float lre = 0.f, lim = 0.f, h_re = 0.f, h_im = 0.f;
    if (tid < HD) {
        float lmod = __expf(-__expf(nu_log[f * HD + tid]));
        float th   = __expf(theta_log[f * HD + tid]);
        lre = lmod * __cosf(th);
        lim = lmod * __sinf(th);
    }
    __syncthreads();

    int buf = 0;
    for (int chunk = 0; chunk < NCHUNK; ++chunk) {
        const int t0 = chunk * TC;

        // ---- wait for this chunk's TDM tile; prefetch next via TDM ----
        if (wave == 0) {
            __builtin_amdgcn_s_wait_tensorcnt(0);
            if (chunk + 1 < NCHUNK)
                tdm_load_x_chunk(xg, b, f, t0 + TC, OFF_XSTG + (buf ^ 1) * STGB);
        }
        __syncthreads();   // staging buffer visible to all waves

        // ---- phase 1: LDS f32 [c][t] -> LDS bf16 [t][c] (transpose + convert) ----
        {
            const float* stg = reinterpret_cast<const float*>(smem + OFF_XSTG + buf * STGB);
            for (int q = tid; q < (TC / 4) * HD; q += NTHREADS) {
                int c = q >> 4;            // channel row in staging
                int t = (q & 15) << 2;     // 4 consecutive t
                float4 v = *reinterpret_cast<const float4*>(stg + c * TC + t);
                xA[(t + 0) * HD + c] = f2bf(v.x);
                xA[(t + 1) * HD + c] = f2bf(v.y);
                xA[(t + 2) * HD + c] = f2bf(v.z);
                xA[(t + 3) * HD + c] = f2bf(v.w);
            }
        }
        __syncthreads();

        // ---- phase 2: u_{re,im}[TC x 128] = xA @ (gamma*B)^T, interleaved chains ----
        for (int tile = wave; tile < 32; tile += NWAVE) {
            const int m0 = (tile >> 3) << 4;
            const int n0 = (tile & 7) << 4;
            f32x8 accR = {0.f, 0.f, 0.f, 0.f, 0.f, 0.f, 0.f, 0.f};
            f32x8 accI = {0.f, 0.f, 0.f, 0.f, 0.f, 0.f, 0.f, 0.f};
#pragma unroll
            for (int k0 = 0; k0 < HD; k0 += 32) {
                bf16x16 a  = ld_tileA(xA,   m0, k0, lane);   // shared A tile
                bf16x16 wr = ld_tileB(wBre, n0, k0, lane);
                bf16x16 wi = ld_tileB(wBim, n0, k0, lane);
                accR = __builtin_amdgcn_wmma_f32_16x16x32_bf16(
                    false, a, false, wr, (short)0, accR, false, false);
                accI = __builtin_amdgcn_wmma_f32_16x16x32_bf16(
                    false, a, false, wi, (short)0, accI, false, false);
            }
            const int n  = n0 + (lane & 15);
            const int hi = lane >> 4;
#pragma unroll
            for (int v = 0; v < 8; ++v) {              // D elem: M = v + 8*hi, N = lane&15
                uRe[(m0 + v + (hi << 3)) * HD + n] = accR[v];
                uIm[(m0 + v + (hi << 3)) * HD + n] = accI[v];
            }
        }
        __syncthreads();

        // ---- phase 3: sequential complex scan h_t = lam*h_{t-1} + u_t ----
        if (tid < HD) {
            for (int t = 0; t < TC; ++t) {
                float ur = uRe[t * HD + tid];
                float ui = uIm[t * HD + tid];
                float nr = fmaf(lre, h_re, fmaf(-lim, h_im, ur));
                float ni = fmaf(lre, h_im, fmaf( lim, h_re, ui));
                h_re = nr; h_im = ni;
                hRe[t * HD + tid] = f2bf(nr);
                hIm[t * HD + tid] = f2bf(-ni);         // fold the minus of h_im @ C_im
            }
        }
        __syncthreads();

        // ---- phase 4: y = hRe@C_re^T + (-hIm)@C_im^T + D*x ; two chains, then add ----
        for (int tile = wave; tile < 32; tile += NWAVE) {
            const int m0 = (tile >> 3) << 4;
            const int h0 = (tile & 7) << 4;
            f32x8 accR = {0.f, 0.f, 0.f, 0.f, 0.f, 0.f, 0.f, 0.f};
            f32x8 accI = {0.f, 0.f, 0.f, 0.f, 0.f, 0.f, 0.f, 0.f};
#pragma unroll
            for (int k0 = 0; k0 < HD; k0 += 32) {
                bf16x16 aR = ld_tileA(hRe,  m0, k0, lane);
                bf16x16 wR = ld_tileB(wCre, h0, k0, lane);
                bf16x16 aI = ld_tileA(hIm,  m0, k0, lane);
                bf16x16 wI = ld_tileB(wCim, h0, k0, lane);
                accR = __builtin_amdgcn_wmma_f32_16x16x32_bf16(
                    false, aR, false, wR, (short)0, accR, false, false);
                accI = __builtin_amdgcn_wmma_f32_16x16x32_bf16(
                    false, aI, false, wI, (short)0, accI, false, false);
            }
            f32x8 acc = accR + accI;
            const int c  = h0 + (lane & 15);
            const int hi = lane >> 4;
            const size_t base = ((size_t)(b * HD + c) * FD + f) * TD + t0 + m0 + (hi << 3);
            const float dd = Dg[f * HD + c];
            float4 x0 = *reinterpret_cast<const float4*>(xg + base);
            float4 x1 = *reinterpret_cast<const float4*>(xg + base + 4);
            float4 o0 = make_float4(acc[0] + dd * x0.x, acc[1] + dd * x0.y,
                                    acc[2] + dd * x0.z, acc[3] + dd * x0.w);
            float4 o1 = make_float4(acc[4] + dd * x1.x, acc[5] + dd * x1.y,
                                    acc[6] + dd * x1.z, acc[7] + dd * x1.w);
            *reinterpret_cast<float4*>(outg + base)     = o0;
            *reinterpret_cast<float4*>(outg + base + 4) = o1;
        }
        __syncthreads();   // hRe/hIm/u reused next chunk
        buf ^= 1;
    }
}

extern "C" void kernel_launch(void* const* d_in, const int* in_sizes, int n_in,
                              void* d_out, int out_size, void* d_ws, size_t ws_size,
                              hipStream_t stream) {
    (void)in_sizes; (void)n_in; (void)out_size; (void)d_ws; (void)ws_size;
    const float* x         = (const float*)d_in[0];
    const float* nu_log    = (const float*)d_in[1];
    const float* theta_log = (const float*)d_in[2];
    const float* gamma_log = (const float*)d_in[3];
    const float* B_re      = (const float*)d_in[4];
    const float* B_im      = (const float*)d_in[5];
    const float* C_re      = (const float*)d_in[6];
    const float* C_im      = (const float*)d_in[7];
    const float* D         = (const float*)d_in[8];
    float* out             = (float*)d_out;

    dim3 grid(FD * BSZ);     // one workgroup per (freq bin, batch)
    dim3 block(NTHREADS);    // 8 wave32
    lru_fused_wmma_kernel<<<grid, block, LDS_BYTES, stream>>>(
        x, nu_log, theta_log, gamma_log, B_re, B_im, C_re, C_im, D, out);
}
